// MambaBlock_50414326120508
// MI455X (gfx1250) — matrix operand order repaired and checked by reading, not compiled, not run on hardware
//
#include <hip/hip_runtime.h>
#include <hip/hip_bf16.h>

// ---------------------------------------------------------------------------
// Problem constants (from the reference)
// ---------------------------------------------------------------------------
#define D_MODEL 2048
#define D_INNER 4096
#define D_CONV  4
#define BATCH   4
#define SEQ     2048
#define BT      (BATCH * SEQ)        // 8192 tokens
#define E2      (2 * D_INNER)        // 8192
#define D_MLP   (4 * D_MODEL)        // 8192

typedef __attribute__((ext_vector_type(16))) __bf16 v16bf;
typedef __attribute__((ext_vector_type(8)))  __bf16 v8bf;
typedef __attribute__((ext_vector_type(8)))  float  v8f;

// ---------------------------------------------------------------------------
// bf16 <-> f32 helpers (bit-level, no reliance on __bf16 arithmetic)
// ---------------------------------------------------------------------------
__device__ __forceinline__ __bf16 to_bf16(float f) {
    union { float f; unsigned u; } v; v.f = f;
    unsigned r = v.u + 0x7FFFu + ((v.u >> 16) & 1u);   // round-to-nearest-even
    unsigned short h = (unsigned short)(r >> 16);
    __bf16 b;
    __builtin_memcpy(&b, &h, 2);
    return b;
}
__device__ __forceinline__ float bf16_to_f(__bf16 b) {
    unsigned short h;
    __builtin_memcpy(&h, &b, 2);
    union { unsigned u; float f; } v; v.u = ((unsigned)h) << 16;
    return v.f;
}
__device__ __forceinline__ float silu(float v) {
    return v / (1.0f + __expf(-v));
}

// ---------------------------------------------------------------------------
// fp32 -> bf16 conversion (4-wide), for weight matrices
// ---------------------------------------------------------------------------
__global__ __launch_bounds__(256) void cvt_f32_bf16_kernel(
    const float* __restrict__ in, __bf16* __restrict__ out, int n4) {
    int i = blockIdx.x * 256 + threadIdx.x;
    if (i >= n4) return;
    float4 v = ((const float4*)in)[i];
    __bf16* o = out + (size_t)i * 4;
    o[0] = to_bf16(v.x); o[1] = to_bf16(v.y);
    o[2] = to_bf16(v.z); o[3] = to_bf16(v.w);
}

// ---------------------------------------------------------------------------
// RMSNorm: h = x / max(||x||, 1e-12) * sqrt(D), one block per token row
// ---------------------------------------------------------------------------
__global__ __launch_bounds__(256) void rmsnorm_bf16_kernel(
    const float* __restrict__ x, __bf16* __restrict__ h) {
    const int row = blockIdx.x;
    const float* px = x + (size_t)row * D_MODEL;
    float s = 0.0f;
    #pragma unroll
    for (int i = threadIdx.x; i < D_MODEL; i += 256) {
        float v = px[i];
        s += v * v;
    }
    __shared__ float sm[256];
    sm[threadIdx.x] = s;
    __syncthreads();
    #pragma unroll
    for (int off = 128; off > 0; off >>= 1) {
        if (threadIdx.x < off) sm[threadIdx.x] += sm[threadIdx.x + off];
        __syncthreads();
    }
    const float nrm   = __builtin_sqrtf(sm[0]);
    const float scale = 45.254833995939045f / fmaxf(nrm, 1e-12f);  // sqrt(2048)
    __bf16* po = h + (size_t)row * D_MODEL;
    #pragma unroll
    for (int i = threadIdx.x; i < D_MODEL; i += 256)
        po[i] = to_bf16(px[i] * scale);
}

// ---------------------------------------------------------------------------
// Fused depthwise causal conv (4 taps) + SiLU gate:
//   y[t,e] = silu(conv_b[e] + sum_j w[e,j]*xp[t-3+j,e]) * silu(z[t,e])
// xz layout: [BT, E2], xp = cols [0,4096), z = cols [4096,8192)
// ---------------------------------------------------------------------------
__global__ __launch_bounds__(256) void conv_gate_kernel(
    const __bf16* __restrict__ xz, const float* __restrict__ conv_w,
    const float* __restrict__ conv_b, __bf16* __restrict__ y) {
    size_t idx = (size_t)blockIdx.x * 256 + threadIdx.x;   // over BT * D_INNER
    if (idx >= (size_t)BT * D_INNER) return;
    const int    e  = (int)(idx & (D_INNER - 1));
    const size_t bt = idx >> 12;                           // idx / 4096
    const int    t  = (int)(bt & (SEQ - 1));

    float acc = conv_b[e];
    const float* w = conv_w + (size_t)e * D_CONV;
    #pragma unroll
    for (int j = 0; j < D_CONV; ++j) {
        int back = (D_CONV - 1) - j;                       // 3,2,1,0
        if (t - back >= 0)
            acc += w[j] * bf16_to_f(xz[(bt - back) * E2 + e]);
    }
    float z = bf16_to_f(xz[bt * E2 + D_INNER + e]);
    y[bt * D_INNER + e] = to_bf16(silu(acc) * silu(z));
}

// ---------------------------------------------------------------------------
// Tiled bf16 WMMA GEMM (NT): C[M,N] = A[M,K] * B[N,K]^T
//   block = 256 threads (8 waves), tile 128x64, wave tile 32x32 (2x2 WMMA)
//   double-buffered LDS staging; EPI: 0=bf16 store, 1=relu->bf16,
//   2 = f32 store of (resid + acc)
// ---------------------------------------------------------------------------
#define BMT 128
#define BNT 64
#define KBT 32
#define LDA_P 40   // padded row stride (halfs): 32 data + 8 pad
#define LDB_P 40

template <int EPI>
__global__ __launch_bounds__(256) void gemm_bf16_wmma_kernel(
    const __bf16* __restrict__ A, const __bf16* __restrict__ Bm,
    void* __restrict__ Cout, const float* __restrict__ resid,
    int M, int N, int K) {
    __shared__ __bf16 sA[2][BMT * LDA_P];
    __shared__ __bf16 sB[2][BNT * LDB_P];

    const int tid  = threadIdx.x;
    const int lane = tid & 31;
    const int wid  = tid >> 5;
    const int wm   = wid & 3;          // wave row   (0..3) -> 32-row slab
    const int wn   = wid >> 2;         // wave col   (0..1) -> 32-col slab
    const int lrow = lane & 15;
    const int hi16 = lane >> 4;        // 0: lanes 0-15, 1: lanes 16-31

    const int bm = blockIdx.x * BMT;
    const int bn = blockIdx.y * BNT;

    // ---- global tile fetch (regs) ------------------------------------
    uint4 ra0, ra1, rb0;
    const int ar0 = tid >> 2,          ac0 = (tid & 3) * 8;
    const int ar1 = (tid + 256) >> 2,  ac1 = ((tid + 256) & 3) * 8;

    auto gload = [&](int kt) {
        const int kk = kt * KBT;
        ra0 = *(const uint4*)(A  + (size_t)(bm + ar0) * K + kk + ac0);
        ra1 = *(const uint4*)(A  + (size_t)(bm + ar1) * K + kk + ac1);
        rb0 = *(const uint4*)(Bm + (size_t)(bn + ar0) * K + kk + ac0);
    };
    auto sstore = [&](int buf) {
        *(uint4*)&sA[buf][ar0 * LDA_P + ac0] = ra0;
        *(uint4*)&sA[buf][ar1 * LDA_P + ac1] = ra1;
        *(uint4*)&sB[buf][ar0 * LDB_P + ac0] = rb0;
    };

    v8f acc[2][2];
    #pragma unroll
    for (int i = 0; i < 2; ++i)
        #pragma unroll
        for (int j = 0; j < 2; ++j)
            acc[i][j] = (v8f){0.f, 0.f, 0.f, 0.f, 0.f, 0.f, 0.f, 0.f};

    const int KT = K / KBT;
    gload(0);
    sstore(0);
    __syncthreads();

    int cur = 0;
    for (int kt = 0; kt < KT; ++kt) {
        if (kt + 1 < KT) gload(kt + 1);
        if (kt + 2 < KT) {  // gfx1250 global_prefetch of the tile after next
            __builtin_prefetch(A  + (size_t)(bm + ar0) * K + (kt + 2) * KBT, 0, 1);
            __builtin_prefetch(Bm + (size_t)(bn + ar0) * K + (kt + 2) * KBT, 0, 1);
        }

        // ---- fragments from LDS (ISA 16-bit A / B lane layouts) ------
        v16bf af[2], bfr[2];
        #pragma unroll
        for (int i = 0; i < 2; ++i) {
            const __bf16* p = &sA[cur][(wm * 32 + i * 16 + lrow) * LDA_P + hi16 * 8];
            v8bf lo = *(const v8bf*)p;          // K = k0 .. k0+7
            v8bf hi = *(const v8bf*)(p + 16);   // K = k0+16 .. k0+23
            af[i] = __builtin_shufflevector(lo, hi,
                    0, 1, 2, 3, 4, 5, 6, 7, 8, 9, 10, 11, 12, 13, 14, 15);
        }
        #pragma unroll
        for (int j = 0; j < 2; ++j) {
            const __bf16* p = &sB[cur][(wn * 32 + j * 16 + lrow) * LDB_P + hi16 * 8];
            v8bf lo = *(const v8bf*)p;          // K = k0 .. k0+7
            v8bf hi = *(const v8bf*)(p + 16);   // K = k0+16 .. k0+23
            bfr[j] = __builtin_shufflevector(lo, hi,
                    0, 1, 2, 3, 4, 5, 6, 7, 8, 9, 10, 11, 12, 13, 14, 15);
        }

        #pragma unroll
        for (int i = 0; i < 2; ++i)
            #pragma unroll
            for (int j = 0; j < 2; ++j)
                acc[i][j] = __builtin_amdgcn_wmma_f32_16x16x32_bf16(
                    false, af[i], false, bfr[j], (short)0, acc[i][j],
                    false, false);

        if (kt + 1 < KT) sstore(cur ^ 1);
        __syncthreads();
        cur ^= 1;
    }

    // ---- epilogue: C/D 16x16 f32 layout (VGPR r -> M = r + hi16*8) ----
    __bf16* Cb = (__bf16*)Cout;
    float*  Cf = (float*)Cout;
    #pragma unroll
    for (int i = 0; i < 2; ++i) {
        #pragma unroll
        for (int j = 0; j < 2; ++j) {
            const int mb = bm + wm * 32 + i * 16 + hi16 * 8;
            const int nc = bn + wn * 32 + j * 16 + lrow;
            #pragma unroll
            for (int r = 0; r < 8; ++r) {
                const size_t off = (size_t)(mb + r) * N + nc;
                float v = acc[i][j][r];
                if constexpr (EPI == 0) {
                    Cb[off] = to_bf16(v);
                } else if constexpr (EPI == 1) {
                    Cb[off] = to_bf16(v > 0.f ? v : 0.f);
                } else {
                    Cf[off] = resid[off] + v;
                }
            }
        }
    }
}

// ---------------------------------------------------------------------------
// Launcher
// ---------------------------------------------------------------------------
extern "C" void kernel_launch(void* const* d_in, const int* in_sizes, int n_in,
                              void* d_out, int out_size, void* d_ws, size_t ws_size,
                              hipStream_t stream) {
    const float* x      = (const float*)d_in[0];   // (B,T,2048)
    const float* W_in   = (const float*)d_in[1];   // (8192,2048)
    const float* conv_w = (const float*)d_in[2];   // (4096,1,4)
    const float* conv_b = (const float*)d_in[3];   // (4096,)
    const float* W_out  = (const float*)d_in[4];   // (2048,4096)
    const float* W_mlp1 = (const float*)d_in[5];   // (8192,2048)
    const float* W_mlp2 = (const float*)d_in[6];   // (2048,8192)
    float* out = (float*)d_out;                    // (B,T,2048) fp32

    // ---- workspace layout (bytes) -----------------------------------
    char* ws = (char*)d_ws;
    const size_t SZ_WIN  = (size_t)E2 * D_MODEL * 2;       // 32 MiB
    const size_t SZ_WOUT = (size_t)D_MODEL * D_INNER * 2;  // 16 MiB
    const size_t SZ_WM1  = (size_t)D_MLP * D_MODEL * 2;    // 32 MiB
    const size_t SZ_WM2  = (size_t)D_MODEL * D_MLP * 2;    // 32 MiB
    const size_t SZ_H    = (size_t)BT * D_MODEL * 2;       // 32 MiB
    const size_t SZ_XZ   = (size_t)BT * E2 * 2;            // 128 MiB (xz, later relu(mlp1))
    __bf16* Wb_in  = (__bf16*)(ws);
    __bf16* Wb_out = (__bf16*)(ws + SZ_WIN);
    __bf16* Wb_m1  = (__bf16*)(ws + SZ_WIN + SZ_WOUT);
    __bf16* Wb_m2  = (__bf16*)(ws + SZ_WIN + SZ_WOUT + SZ_WM1);
    __bf16* hb     = (__bf16*)(ws + SZ_WIN + SZ_WOUT + SZ_WM1 + SZ_WM2);
    __bf16* xzb    = (__bf16*)(ws + SZ_WIN + SZ_WOUT + SZ_WM1 + SZ_WM2 + SZ_H);
    __bf16* yb     = (__bf16*)(ws + SZ_WIN + SZ_WOUT + SZ_WM1 + SZ_WM2 + SZ_H + SZ_XZ);

    dim3 blk(256);

    // 1) weights -> bf16
    {
        int n4;
        n4 = (E2 * D_MODEL) / 4;
        cvt_f32_bf16_kernel<<<(n4 + 255) / 256, blk, 0, stream>>>(W_in,   Wb_in,  n4);
        n4 = (D_MODEL * D_INNER) / 4;
        cvt_f32_bf16_kernel<<<(n4 + 255) / 256, blk, 0, stream>>>(W_out,  Wb_out, n4);
        n4 = (D_MLP * D_MODEL) / 4;
        cvt_f32_bf16_kernel<<<(n4 + 255) / 256, blk, 0, stream>>>(W_mlp1, Wb_m1,  n4);
        n4 = (D_MODEL * D_MLP) / 4;
        cvt_f32_bf16_kernel<<<(n4 + 255) / 256, blk, 0, stream>>>(W_mlp2, Wb_m2,  n4);
    }

    // 2) h = rmsnorm(x) -> bf16
    rmsnorm_bf16_kernel<<<BT, blk, 0, stream>>>(x, hb);

    // 3) xz = h @ W_in^T          [8192 x 8192], K=2048
    gemm_bf16_wmma_kernel<0><<<dim3(BT / BMT, E2 / BNT), blk, 0, stream>>>(
        hb, Wb_in, xzb, nullptr, BT, E2, D_MODEL);

    // 4) y = silu(causal_conv(xp)+b) * silu(z) -> bf16   [8192 x 4096]
    {
        size_t tot = (size_t)BT * D_INNER;
        conv_gate_kernel<<<(unsigned)((tot + 255) / 256), blk, 0, stream>>>(
            xzb, conv_w, conv_b, yb);
    }

    // 5) out = x + y @ W_out^T    [8192 x 2048], K=4096  (residual in epilogue)
    gemm_bf16_wmma_kernel<2><<<dim3(BT / BMT, D_MODEL / BNT), blk, 0, stream>>>(
        yb, Wb_out, out, x, BT, D_MODEL, D_INNER);

    // 6) h2 = rmsnorm(out) -> bf16
    rmsnorm_bf16_kernel<<<BT, blk, 0, stream>>>(out, hb);

    // 7) u = relu(h2 @ W_mlp1^T)  [8192 x 8192], K=2048  (reuse xz buffer)
    gemm_bf16_wmma_kernel<1><<<dim3(BT / BMT, D_MLP / BNT), blk, 0, stream>>>(
        hb, Wb_m1, xzb, nullptr, BT, D_MLP, D_MODEL);

    // 8) out = out + u @ W_mlp2^T [8192 x 2048], K=8192  (residual = out itself)
    gemm_bf16_wmma_kernel<2><<<dim3(BT / BMT, D_MODEL / BNT), blk, 0, stream>>>(
        xzb, Wb_m2, out, out, BT, D_MODEL, D_MLP);
}
